// Sggnn_all_50259707298049
// MI455X (gfx1250) — matrix operand order repaired and checked by reading, not compiled, not run on hardware
//
#include <hip/hip_runtime.h>

#define CC 512
#define GG 1024
#define PP 128
#define TM 128
#define XLS 520   // bf16 elems per LDS X row (512 + 8 pad)
#define WPS 40    // bf16 elems per LDS W-panel row (32 + 8 pad)
#define WPBUF (CC * WPS)   // one panel buffer in bf16 elems

typedef __attribute__((ext_vector_type(16))) __bf16 v16bf;
typedef __attribute__((ext_vector_type(8)))  float  v8f;
typedef __attribute__((ext_vector_type(4)))  float  f32x4;

__device__ __forceinline__ unsigned short f2bf(float x) {
  union { float f; unsigned int u; } v; v.f = x;
  unsigned int u = v.u;
  unsigned int r = u + 0x7fffu + ((u >> 16) & 1u);   // round-to-nearest-even
  return (unsigned short)(r >> 16);
}

// pack 4 floats -> 4 bf16 and store 8B into LDS X tile at (row, slab*32 + c*4)
__device__ __forceinline__ void stx(unsigned short* __restrict__ Xl,
                                    int row, int slab, int c, f32x4 f) {
  unsigned long long pk =
      (unsigned long long)((unsigned)f2bf(f.x) | ((unsigned)f2bf(f.y) << 16)) |
      ((unsigned long long)((unsigned)f2bf(f.z) | ((unsigned)f2bf(f.w) << 16)) << 32);
  *(unsigned long long*)&Xl[row * XLS + slab * 32 + c * 4] = pk;
}

// ---------------- prep: class counts -> dinv, BN fold -> (s,t) ----------------
__global__ void prep_scalars(const int* __restrict__ yg,
                             const float* __restrict__ b1, const float* __restrict__ g1,
                             const float* __restrict__ be1, const float* __restrict__ rm1,
                             const float* __restrict__ rv1,
                             const float* __restrict__ b2, const float* __restrict__ g2,
                             const float* __restrict__ be2, const float* __restrict__ rm2,
                             const float* __restrict__ rv2,
                             float* __restrict__ dinv,
                             float* __restrict__ s1, float* __restrict__ t1,
                             float* __restrict__ s2, float* __restrict__ t2) {
  __shared__ int cnt[64];
  int t = threadIdx.x;                 // block of 1024
  if (t < 64) cnt[t] = 0;
  __syncthreads();
  atomicAdd(&cnt[yg[t]], 1);
  __syncthreads();
  dinv[t] = rsqrtf((float)cnt[yg[t]] + 1.0f);   // rowsum = count(class) + 1 (self loop)
  if (t < CC) {
    float i1 = rsqrtf(rv1[t] + 1e-5f);
    float ss1 = g1[t] * i1;
    s1[t] = ss1; t1[t] = (b1[t] - rm1[t]) * ss1 + be1[t];
    float i2 = rsqrtf(rv2[t] + 1e-5f);
    float ss2 = g2[t] * i2;
    s2[t] = ss2; t2[t] = (b2[t] - rm2[t]) * ss2 + be2[t];
  }
}

// ---------------- prep: transpose + fp32 -> bf16 for W1, W2 ----------------
__global__ void prep_transpose(const float* __restrict__ W1, const float* __restrict__ W2,
                               unsigned short* __restrict__ W1t, unsigned short* __restrict__ W2t) {
  int idx = blockIdx.x * 256 + threadIdx.x;     // 0 .. 2*512*512-1
  int which = idx >> 18;
  int r = idx & 262143;
  int k = r >> 9, n = r & 511;
  const float* W = which ? W2 : W1;
  unsigned short* Wt = which ? W2t : W1t;
  Wt[n * CC + k] = f2bf(W[k * CC + n]);         // Wt[n][k] = W[k][n]
}

// ---- core: 512-K GEMM, double-buffered LDS W-panels + reg prefetch; optionally
// ---- streams the X tile from HBM slab-by-slab, overlapped with the k-loop ----
template<bool STREAM>
__device__ __forceinline__ void gemm512(unsigned short* __restrict__ Xl,
                                        unsigned short* __restrict__ Wp,   // 2 * WPBUF
                                        const unsigned short* __restrict__ Wt,
                                        const f32x4* __restrict__ dsrc,    // X tile base (float4), STREAM only
                                        int tid, int m_t, int nh, int lm, int hi,
                                        v8f acc[16]) {
#pragma unroll
  for (int nt = 0; nt < 16; ++nt) {
    v8f z = {0.f, 0.f, 0.f, 0.f, 0.f, 0.f, 0.f, 0.f};
    acc[nt] = z;
  }
  // ---- W staging: thread tid owns channel row 'tid'; panel row = 32 bf16 = 4 x uint4
  const uint4* wsrc  = (const uint4*)(Wt + (size_t)tid * CC);       // 64 uint4 per channel row
  uint4* wdst0 = (uint4*)(Wp + tid * WPS);
  uint4* wdst1 = (uint4*)(Wp + WPBUF + tid * WPS);
  {
    uint4 c0 = wsrc[0], c1 = wsrc[1], c2 = wsrc[2], c3 = wsrc[3];
    wdst0[0] = c0; wdst0[1] = c1; wdst0[2] = c2; wdst0[3] = c3;
  }
  uint4 p0 = wsrc[4], p1 = wsrc[5], p2 = wsrc[6], p3 = wsrc[7];

  // ---- X streaming: slab s = K cols [s*32, s*32+32); per thread 2 float4:
  //      linear q ∈ {tid, tid+512}: row = q>>3 (0..127), c = q&7
  int r0 = tid >> 3, c0 = tid & 7, r1 = (tid + 512) >> 3;
  f32x4 xa, xb;
  if (STREAM) {
    xa = __builtin_nontemporal_load(&dsrc[r0 * 128 + c0]);          // slab 0
    xb = __builtin_nontemporal_load(&dsrc[r1 * 128 + c0]);
    stx(Xl, r0, 0, c0, xa);
    stx(Xl, r1, 0, c0, xb);
    xa = __builtin_nontemporal_load(&dsrc[r0 * 128 + 8 + c0]);      // prefetch slab 1
    xb = __builtin_nontemporal_load(&dsrc[r1 * 128 + 8 + c0]);
  }
  __syncthreads();                                   // slab0 + W buf0 visible

  for (int kp = 0; kp < CC; kp += 32) {
    const int i = kp >> 5;                           // panel/slab index 0..15
    // store prefetched W panel (i+1) into the buffer not read this iteration
    if (i + 1 < 16) {
      uint4* dst = (i & 1) ? wdst0 : wdst1;
      dst[0] = p0; dst[1] = p1; dst[2] = p2; dst[3] = p3;
    }
    // prefetch W panel (i+2) from global (L2-resident)
    if (i + 2 < 16) {
      p0 = wsrc[(i + 2) * 4 + 0];
      p1 = wsrc[(i + 2) * 4 + 1];
      p2 = wsrc[(i + 2) * 4 + 2];
      p3 = wsrc[(i + 2) * 4 + 3];
    }
    if (STREAM) {
      // store X slab (i+1) (read only after next barrier), prefetch slab (i+2) from HBM
      if (i + 1 < 16) {
        stx(Xl, r0, i + 1, c0, xa);
        stx(Xl, r1, i + 1, c0, xb);
      }
      if (i + 2 < 16) {
        xa = __builtin_nontemporal_load(&dsrc[r0 * 128 + (i + 2) * 8 + c0]);
        xb = __builtin_nontemporal_load(&dsrc[r1 * 128 + (i + 2) * 8 + c0]);
      }
    }
    const unsigned short* buf = Wp + (i & 1) * WPBUF;

    // A fragment: M = m_t*16 + lm, elems 0..15 -> K=kp+hi*16+s
    union { v16bf v; uint4 q[2]; } A;
    const unsigned short* ar = Xl + (m_t * 16 + lm) * XLS + kp + hi * 16;
    A.q[0] = *(const uint4*)ar;
    A.q[1] = *(const uint4*)(ar + 8);

    // B fragments, explicit 2-deep pipeline: load nt+1 before issuing WMMA nt
    union { v16bf v; uint4 q[2]; } B[2];
    const unsigned short* brow = buf + (nh * 256 + lm) * WPS + hi * 16;
    B[0].q[0] = *(const uint4*)brow;
    B[0].q[1] = *(const uint4*)(brow + 8);
#pragma unroll
    for (int nt = 0; nt < 16; ++nt) {
      if (nt + 1 < 16) {
        const unsigned short* bn = brow + (nt + 1) * (16 * WPS);
        B[(nt + 1) & 1].q[0] = *(const uint4*)bn;
        B[(nt + 1) & 1].q[1] = *(const uint4*)(bn + 8);
      }
      acc[nt] = __builtin_amdgcn_wmma_f32_16x16x32_bf16(
          false, A.v, false, B[nt & 1].v, (short)0, acc[nt], false, false);
    }
    __syncthreads();   // this iteration's reads done; writes above were to other regions
  }
}

// ---------------- fused: (Lin+BN+LReLU) x2 then row-dot with Wc -> u[row][2] ----------------
__global__ __launch_bounds__(512)
void fused_mlp(const float* __restrict__ d,
               const unsigned short* __restrict__ W1t,
               const unsigned short* __restrict__ W2t,
               const float* __restrict__ s1, const float* __restrict__ t1,
               const float* __restrict__ s2, const float* __restrict__ t2,
               const float* __restrict__ Wc,
               float* __restrict__ u) {
  __shared__ unsigned short Xl[TM * XLS];    // 133,120 B : X tile then Y1 tile (bf16)
  __shared__ unsigned short Wp[2 * WPBUF];   //  81,920 B : double-buffered W k-panels
  __shared__ float u_l[TM * 2];              //   1,024 B

  const int tid = threadIdx.x;
  const int wave = tid >> 5, lane = tid & 31;
  const int m_t = wave & 7, nh = wave >> 3;      // 8 M-tiles x 2 N-halves
  const int lm = lane & 15, hi = lane >> 4;
  const size_t rowbase = (size_t)blockIdx.x * TM;

  if (tid < TM * 2) u_l[tid] = 0.f;

  v8f acc[16];

  // Layer 1: X tile streamed from HBM inside the k-loop (load/compute overlap)
  const f32x4* dsrc = (const f32x4*)(d + rowbase * CC);
  gemm512<true>(Xl, Wp, W1t, dsrc, tid, m_t, nh, lm, hi, acc);

  __syncthreads();                                // everyone done reading Xl
#pragma unroll
  for (int nt = 0; nt < 16; ++nt) {
    int c = nh * 256 + nt * 16 + lm;
    float s = s1[c], t = t1[c];
#pragma unroll
    for (int r = 0; r < 8; ++r) {
      float y = acc[nt][r] * s + t;               // BN(eval) folded affine
      y = y >= 0.f ? y : 0.1f * y;                // LeakyReLU(0.1)
      Xl[(m_t * 16 + hi * 8 + r) * XLS + c] = f2bf(y);
    }
  }
  __syncthreads();

  // Layer 2 (Y1 already resident in LDS)
  gemm512<false>(Xl, Wp, W2t, (const f32x4*)nullptr, tid, m_t, nh, lm, hi, acc);

  // Fold classifier: u[row][j] = t_row . Wc[:,j]   (t never materialized)
  float pu0[8], pu1[8];
#pragma unroll
  for (int r = 0; r < 8; ++r) { pu0[r] = 0.f; pu1[r] = 0.f; }
#pragma unroll
  for (int nt = 0; nt < 16; ++nt) {
    int c = nh * 256 + nt * 16 + lm;
    float s = s2[c], t = t2[c];
    float w0 = Wc[2 * c], w1 = Wc[2 * c + 1];
#pragma unroll
    for (int r = 0; r < 8; ++r) {
      float y = acc[nt][r] * s + t;
      y = y >= 0.f ? y : 0.1f * y;
      pu0[r] += y * w0;
      pu1[r] += y * w1;
    }
  }
#pragma unroll
  for (int r = 0; r < 8; ++r) {
    int row = m_t * 16 + hi * 8 + r;
    atomicAdd(&u_l[row * 2 + 0], pu0[r]);
    atomicAdd(&u_l[row * 2 + 1], pu1[r]);
  }
  __syncthreads();
  if (tid < TM * 2) u[rowbase * 2 + tid] = u_l[tid];
}

// ---------------- finalize: class sums of u, then combine ----------------
__global__ void finalize_k(const float* __restrict__ u, const float* __restrict__ dinv,
                           const int* __restrict__ yg, const float* __restrict__ bc,
                           float* __restrict__ out) {
  __shared__ float vl[128];                       // [64 classes][2]
  int t = threadIdx.x, p = blockIdx.x;
  if (t < 128) vl[t] = 0.f;
  __syncthreads();
  for (int g = t; g < GG; g += 256) {
    float w = dinv[g]; int k = yg[g];
    const float* up = u + ((size_t)p * GG + g) * 2;
    atomicAdd(&vl[2 * k + 0], w * up[0]);
    atomicAdd(&vl[2 * k + 1], w * up[1]);
  }
  __syncthreads();
  float bc0 = bc[0], bc1 = bc[1];
  for (int h = t; h < GG; h += 256) {
    float di = dinv[h]; int k = yg[h];
    size_t base = ((size_t)p * GG + h) * 2;
    out[base + 0] = di * vl[2 * k + 0] + di * di * u[base + 0] + bc0;
    out[base + 1] = di * vl[2 * k + 1] + di * di * u[base + 1] + bc1;
  }
}

// ---------------- labels ----------------
__global__ void labels_k(const int* __restrict__ yp, const int* __restrict__ yg,
                         float* __restrict__ out) {
  int idx = blockIdx.x * 256 + threadIdx.x;       // 0 .. P*G-1
  int i = idx >> 10, j = idx & 1023;
  out[idx] = (yp[i] == yg[j]) ? 1.0f : 0.0f;
}

extern "C" void kernel_launch(void* const* d_in, const int* in_sizes, int n_in,
                              void* d_out, int out_size, void* d_ws, size_t ws_size,
                              hipStream_t stream) {
  const float* d   = (const float*)d_in[0];
  const int*   y_p = (const int*)d_in[1];
  const int*   y_g = (const int*)d_in[2];
  const float* W1  = (const float*)d_in[3];
  const float* b1  = (const float*)d_in[4];
  const float* g1  = (const float*)d_in[5];
  const float* be1 = (const float*)d_in[6];
  const float* rm1 = (const float*)d_in[7];
  const float* rv1 = (const float*)d_in[8];
  const float* W2  = (const float*)d_in[9];
  const float* b2  = (const float*)d_in[10];
  const float* g2  = (const float*)d_in[11];
  const float* be2 = (const float*)d_in[12];
  const float* rm2 = (const float*)d_in[13];
  const float* rv2 = (const float*)d_in[14];
  const float* Wc  = (const float*)d_in[15];
  const float* bc  = (const float*)d_in[16];

  char* ws = (char*)d_ws;
  unsigned short* W1t = (unsigned short*)(ws + 0);        // 512 KB
  unsigned short* W2t = (unsigned short*)(ws + 524288);   // 512 KB
  float* dinv = (float*)(ws + 1048576);                   //   4 KB
  float* s1   = (float*)(ws + 1052672);                   //   2 KB
  float* t1   = (float*)(ws + 1054720);
  float* s2   = (float*)(ws + 1056768);
  float* t2   = (float*)(ws + 1058816);
  float* u    = (float*)(ws + 1060864);                   //   1 MB

  float* out_res = (float*)d_out;                         // [P*G][2]
  float* out_lab = out_res + (size_t)PP * GG * 2;         // [P*G]

  hipLaunchKernelGGL(prep_scalars, dim3(1), dim3(1024), 0, stream,
                     y_g, b1, g1, be1, rm1, rv1, b2, g2, be2, rm2, rv2,
                     dinv, s1, t1, s2, t2);
  hipLaunchKernelGGL(prep_transpose, dim3(2048), dim3(256), 0, stream, W1, W2, W1t, W2t);
  hipLaunchKernelGGL(fused_mlp, dim3(1024), dim3(512), 0, stream,
                     d, W1t, W2t, s1, t1, s2, t2, Wc, u);
  hipLaunchKernelGGL(finalize_k, dim3(PP), dim3(256), 0, stream, u, dinv, y_g, bc, out_res);
  hipLaunchKernelGGL(labels_k, dim3((PP * GG) / 256), dim3(256), 0, stream, y_p, y_g, out_lab);
}
